// PAM_Module_50646254355110
// MI455X (gfx1250) — matrix operand order repaired and checked by reading, not compiled, not run on hardware
//
#include <hip/hip_runtime.h>
#include <stdint.h>
#include <stddef.h>

// ---------------------------------------------------------------- types ----
typedef __bf16 bf16_t;
typedef __attribute__((ext_vector_type(16))) __bf16 v16bf;
typedef __attribute__((ext_vector_type(8)))  __bf16 v8bf;
typedef __attribute__((ext_vector_type(8)))  float  v8f;
typedef __attribute__((ext_vector_type(4)))  float  v4f;

#define B_   8
#define C_   512
#define N_   4096
#define KD_  64
#define OD_  512
#define WR_  640   // 64 + 64 + 512 packed weight rows

__device__ __forceinline__ bf16_t f2bf(float f) {
    union { float f; uint32_t u; } v; v.f = f;
    uint32_t r = v.u + 0x7FFFu + ((v.u >> 16) & 1u);   // round-to-nearest-even
    uint16_t h = (uint16_t)(r >> 16);
    bf16_t out; __builtin_memcpy(&out, &h, 2);
    return out;
}

__device__ __forceinline__ v16bf cat8(v8bf lo, v8bf hi) {
    return __builtin_shufflevector(lo, hi, 0,1,2,3,4,5,6,7,8,9,10,11,12,13,14,15);
}

// ------------------------------------------------- kernel 1: pack weights --
__global__ void pam_prep_w(const float* __restrict__ qw, const float* __restrict__ qb,
                           const float* __restrict__ kw, const float* __restrict__ kb,
                           const float* __restrict__ vw, const float* __restrict__ vb,
                           bf16_t* __restrict__ Wbf, float* __restrict__ bias) {
    int idx = blockIdx.x * 256 + threadIdx.x;         // over 640*512
    if (idx < WR_ * C_) {
        int r = idx / C_, c = idx % C_;
        float val;
        if (r < 64)       val = qw[r * C_ + c];
        else if (r < 128) val = kw[(r - 64) * C_ + c];
        else              val = vw[(r - 128) * C_ + c];
        Wbf[idx] = f2bf(val);
    }
    if (idx < WR_) {
        float bv;
        if (idx < 64)       bv = qb[idx];
        else if (idx < 128) bv = kb[idx - 64];
        else                bv = vb[idx - 128];
        bias[idx] = bv;
    }
}

// ------------------------------------- kernel 2: x [B,C,N] -> XT[B,N,C] bf16
__global__ void pam_xpose(const float* __restrict__ x, bf16_t* __restrict__ xt) {
    long idx = (long)blockIdx.x * 256 + threadIdx.x;  // over B*N*C
    int  c = (int)(idx % C_);
    long t = idx / C_;
    int  n = (int)(t % N_);
    int  b = (int)(t / N_);
    xt[idx] = f2bf(x[((long)(b * C_ + c)) * N_ + n]);
}

// ---------------------- kernel 3: QKV = W(bf16) x XT^T, one wave per tile --
// Q,K -> [B][N][64] bf16 ; V -> [B][OD][N] bf16 + f32 residual into out.
__global__ __launch_bounds__(256)
void pam_qkv(const bf16_t* __restrict__ xt, const bf16_t* __restrict__ Wbf,
             const float* __restrict__ bias,
             bf16_t* __restrict__ Qbf, bf16_t* __restrict__ Kbf,
             bf16_t* __restrict__ Vbf, float* __restrict__ outv) {
    int wave = blockIdx.x * 8 + (threadIdx.x >> 5);   // 81920 waves total
    int lane = threadIdx.x & 31;
    int lm = lane & 15, half = lane >> 4;
    int nt = wave % 256;
    int t  = wave / 256;
    int mt = t % 40;
    int b  = t / 40;

    v8f acc = {};
    const bf16_t* wrow = Wbf + (size_t)(mt * 16 + lm) * C_;
    const bf16_t* xrow = xt  + ((long)(b * N_ + nt * 16 + lm)) * C_;
    #pragma unroll
    for (int cb = 0; cb < 16; ++cb) {
        int c0 = cb * 32 + half * 8;
        v8bf alo = *(const v8bf*)(wrow + c0);
        v8bf ahi = *(const v8bf*)(wrow + c0 + 16);
        v16bf a  = cat8(alo, ahi);
        v16bf bm = cat8(*(const v8bf*)(xrow + c0), *(const v8bf*)(xrow + c0 + 16));
        acc = __builtin_amdgcn_wmma_f32_16x16x32_bf16(false, a, false, bm,
                                                      (short)0, acc, false, false);
    }
    int n = nt * 16 + lm;
    #pragma unroll
    for (int r = 0; r < 8; ++r) {
        int m = mt * 16 + r + half * 8;
        float val = acc[r] + bias[m];
        if (m < 64) {
            Qbf[((long)(b * N_ + n)) * KD_ + m] = f2bf(val);
        } else if (m < 128) {
            Kbf[((long)(b * N_ + n)) * KD_ + (m - 64)] = f2bf(val);
        } else {
            long oidx = ((long)(b * OD_ + (m - 128))) * N_ + n;
            Vbf[oidx]  = f2bf(val);
            outv[oidx] = val;          // f32 residual V
        }
    }
}

// ------------------- kernel 4: fused flash attention + gamma*O/l + V -------
// grid: B * (N/32) blocks, 256 threads (8 waves). Score panel per iteration:
// 32 queries x 64 keys = 8 sixteen-square tiles, one per wave (fully balanced).
// Every wave also owns OD columns [wave*64, wave*64+64): acc = 2x4 f32 tiles.
__global__ __launch_bounds__(256)
void pam_attn(const bf16_t* __restrict__ Qbf, const bf16_t* __restrict__ Kbf,
              const bf16_t* __restrict__ Vbf, const float* __restrict__ gamma,
              float* __restrict__ out) {
    __shared__ bf16_t Pbuf[32 * 64];        // exp'd scores, bf16, row-major
    __shared__ float  pmaxBuf[16][8];       // [row_in_tile][tile]  tile = rt*4+jt
    __shared__ float  psumBuf[16][8];       // [row_in_tile][tile]

    int wave = threadIdx.x >> 5;
    int lane = threadIdx.x & 31;
    int lm = lane & 15, half = lane >> 4;
    int b     = blockIdx.x >> 7;            // 128 i-tiles per batch
    int ibase = (blockIdx.x & 127) * 32;
    int rts = wave >> 2;                    // this wave's S row-tile (0..1)
    int jts = wave & 3;                     // this wave's S col-tile (0..3)
    int rit = half * 8;                     // row_in_tile base for this half

    v8f acc[2][4] = {};
    float mI[2][8], lI[2][8];
    #pragma unroll
    for (int rt = 0; rt < 2; ++rt)
        #pragma unroll
        for (int r = 0; r < 8; ++r) { mI[rt][r] = -3.0e38f; lI[rt][r] = 0.0f; }

    // Q A-tile for this wave's S row-tile (both K-chunks of kd=64)
    v16bf qa[2];
    {
        const bf16_t* qrow = Qbf + ((long)(b * N_ + ibase + rts * 16 + lm)) * KD_;
        #pragma unroll
        for (int kc = 0; kc < 2; ++kc) {
            int c0 = kc * 32 + half * 8;
            v8bf lo = *(const v8bf*)(qrow + c0);
            v8bf hi = *(const v8bf*)(qrow + c0 + 16);
            qa[kc] = cat8(lo, hi);
        }
    }

    for (int jb = 0; jb < N_; jb += 64) {
        // ---- my 16x16 S tile: rows ibase+rts*16.., cols jb+jts*16.. ----
        v8f s = {};
        {
            const bf16_t* krow = Kbf + ((long)(b * N_ + jb + jts * 16 + lm)) * KD_;
            v16bf kb0 = cat8(*(const v8bf*)(krow + half * 8), *(const v8bf*)(krow + half * 8 + 16));
            v16bf kb1 = cat8(*(const v8bf*)(krow + 32 + half * 8), *(const v8bf*)(krow + 32 + half * 8 + 16));
            s = __builtin_amdgcn_wmma_f32_16x16x32_bf16(false, qa[0], false, kb0,
                                                        (short)0, s, false, false);
            s = __builtin_amdgcn_wmma_f32_16x16x32_bf16(false, qa[1], false, kb1,
                                                        (short)0, s, false, false);
        }
        // per-row tile max (16-lane reduction stays inside the half)
        #pragma unroll
        for (int r = 0; r < 8; ++r) {
            float mx = s[r];
            #pragma unroll
            for (int m = 1; m < 16; m <<= 1) mx = fmaxf(mx, __shfl_xor(mx, m, 32));
            if (lm == 0) pmaxBuf[r + rit][wave] = mx;
        }
        __syncthreads();   // barrier 1: tile maxima visible

        // combine 4 column-partials -> new max + rescale factor (all waves,
        // redundantly but bit-identically)
        float esc[2][8], nm[2][8];
        #pragma unroll
        for (int rt = 0; rt < 2; ++rt) {
            #pragma unroll
            for (int r = 0; r < 8; ++r) {
                v4f pm = *(const v4f*)&pmaxBuf[r + rit][rt * 4];
                float gm = fmaxf(fmaxf(pm.x, pm.y), fmaxf(pm.z, pm.w));
                float nv = fmaxf(mI[rt][r], gm);
                esc[rt][r] = __expf(mI[rt][r] - nv);
                nm[rt][r]  = nv;
            }
        }
        // owning wave writes P = exp(S - m_new) and its row partial sums
        #pragma unroll
        for (int r = 0; r < 8; ++r) {
            float p = __expf(s[r] - nm[rts][r]);
            Pbuf[(rts * 16 + r + rit) * 64 + jts * 16 + lm] = f2bf(p);
            float su = p;
            #pragma unroll
            for (int m = 1; m < 16; m <<= 1) su += __shfl_xor(su, m, 32);
            if (lm == 0) psumBuf[r + rit][wave] = su;
        }
        __syncthreads();   // barrier 2: P + partial sums visible

        // update l/m, rescale accumulators
        #pragma unroll
        for (int rt = 0; rt < 2; ++rt) {
            #pragma unroll
            for (int r = 0; r < 8; ++r) {
                v4f ps = *(const v4f*)&psumBuf[r + rit][rt * 4];
                lI[rt][r] = lI[rt][r] * esc[rt][r] + (ps.x + ps.y + ps.z + ps.w);
                mI[rt][r] = nm[rt][r];
            }
        }
        #pragma unroll
        for (int rt = 0; rt < 2; ++rt)
            #pragma unroll
            for (int ct = 0; ct < 4; ++ct)
                #pragma unroll
                for (int r = 0; r < 8; ++r)
                    acc[rt][ct][r] *= esc[rt][r];

        // P A-tiles (2 row-tiles x 2 j-chunks) from LDS
        v16bf pa[2][2];
        #pragma unroll
        for (int rt = 0; rt < 2; ++rt) {
            const bf16_t* prow = Pbuf + (rt * 16 + lm) * 64;
            #pragma unroll
            for (int kc = 0; kc < 2; ++kc) {
                v8bf lo = *(const v8bf*)(prow + kc * 32 + half * 8);
                v8bf hi = *(const v8bf*)(prow + kc * 32 + half * 8 + 16);
                pa[rt][kc] = cat8(lo, hi);
            }
        }
        // O += P * V  (V B-tiles streamed from L2-resident [B][OD][N])
        #pragma unroll
        for (int kc = 0; kc < 2; ++kc) {
            #pragma unroll
            for (int ct = 0; ct < 4; ++ct) {
                const bf16_t* vrow = Vbf + ((long)(b * OD_ + wave * 64 + ct * 16 + lm)) * N_
                                         + jb + kc * 32 + half * 8;
                v16bf bv = cat8(*(const v8bf*)vrow, *(const v8bf*)(vrow + 16));
                #pragma unroll
                for (int rt = 0; rt < 2; ++rt)
                    acc[rt][ct] = __builtin_amdgcn_wmma_f32_16x16x32_bf16(false, pa[rt][kc],
                                                                          false, bv, (short)0,
                                                                          acc[rt][ct],
                                                                          false, false);
            }
        }
        // no trailing barrier needed: next iteration's LDS writes happen only
        // after barrier 1 / barrier 2 of that iteration, and all reads of this
        // iteration's buffers precede those barriers in program order.
    }

    // epilogue: out = gamma * O / l + V   (every wave has its own final l)
    float g = gamma[0];
    #pragma unroll
    for (int rt = 0; rt < 2; ++rt) {
        #pragma unroll
        for (int r = 0; r < 8; ++r) {
            int row = rt * 16 + r + rit;
            float inv = 1.0f / lI[rt][r];
            int i = ibase + row;
            #pragma unroll
            for (int ct = 0; ct < 4; ++ct) {
                int o = wave * 64 + ct * 16 + lm;
                long idx = ((long)(b * OD_ + o)) * N_ + i;
                out[idx] = g * acc[rt][ct][r] * inv + out[idx];   // + V residual
            }
        }
    }
}

// --------------------------------------------------------------- launch ----
extern "C" void kernel_launch(void* const* d_in, const int* in_sizes, int n_in,
                              void* d_out, int out_size, void* d_ws, size_t ws_size,
                              hipStream_t stream) {
    const float* x     = (const float*)d_in[0];
    const float* qw    = (const float*)d_in[1];
    const float* qb    = (const float*)d_in[2];
    const float* kw    = (const float*)d_in[3];
    const float* kb    = (const float*)d_in[4];
    const float* vw    = (const float*)d_in[5];
    const float* vb    = (const float*)d_in[6];
    const float* gamma = (const float*)d_in[7];
    float* out = (float*)d_out;

    char* ws = (char*)d_ws;
    size_t off = 0;
    bf16_t* XT  = (bf16_t*)(ws + off); off += (size_t)B_ * N_ * C_ * 2;   // 32 MB
    bf16_t* Wbf = (bf16_t*)(ws + off); off += (size_t)WR_ * C_ * 2;       // 640 KB
    float*  bia = (float*)(ws + off);  off += (size_t)WR_ * 4;
    off = (off + 255) & ~(size_t)255;
    bf16_t* Qbf = (bf16_t*)(ws + off); off += (size_t)B_ * N_ * KD_ * 2;  // 4 MB
    bf16_t* Kbf = (bf16_t*)(ws + off); off += (size_t)B_ * N_ * KD_ * 2;  // 4 MB
    bf16_t* Vbf = (bf16_t*)(ws + off); off += (size_t)B_ * OD_ * N_ * 2;  // 32 MB

    pam_prep_w<<<(WR_ * C_ + 255) / 256, 256, 0, stream>>>(qw, qb, kw, kb, vw, vb, Wbf, bia);
    pam_xpose <<<(B_ * (long)N_ * C_) / 256, 256, 0, stream>>>(x, XT);
    pam_qkv   <<<(B_ * 40 * 256) / 8, 256, 0, stream>>>(XT, Wbf, bia, Qbf, Kbf, Vbf, out);
    pam_attn  <<<B_ * (N_ / 32), 256, 0, stream>>>(Qbf, Kbf, Vbf, gamma, out);
}
